// SADUBackbone_64862596104596
// MI455X (gfx1250) — matrix operand order repaired and checked by reading, not compiled, not run on hardware
//
#include <hip/hip_runtime.h>
#include <hip/hip_bf16.h>
#include <math.h>

#define EPSV 1e-5f

typedef _Float16 v16h __attribute__((ext_vector_type(16)));
typedef float    v8f  __attribute__((ext_vector_type(8)));

union H16 { v16h v; uint4 q[2]; };

// ---------------------------------------------------------------- kernel A --
// input BN + 3 depthwise/pointwise branches + leaky relu + adaptive max pool
// output: z_h in f16, position-major: z_h[(b*4096+t)*192 + c]
struct AParams {
  const float* x;
  const float *ing, *inb, *inm, *invv;
  const float* dw[3];
  const float* pw[3];
  const float *bg[3], *bbt[3], *bm[3], *bv[3];
};

__global__ __launch_bounds__(256) void kernA(AParams P, _Float16* __restrict__ z_h) {
  __shared__ float xs[16 * 264];          // BN'd input with halo (260 used)
  __shared__ float pw_s[3 * 1024];
  __shared__ float dw_s[256];             // offsets 0 / 48 / 128
  __shared__ float bsc[192], bsh[192];    // folded branch BN
  __shared__ float isc[16], ish[16];      // folded input BN

  const int tid = threadIdx.x;
  const int b   = blockIdx.x >> 6;
  const int t0  = (blockIdx.x & 63) * 64;

  for (int i = tid; i < 1024; i += 256) {
    pw_s[i]        = P.pw[0][i];
    pw_s[1024 + i] = P.pw[1][i];
    pw_s[2048 + i] = P.pw[2][i];
  }
  if (tid < 48)  dw_s[tid]       = P.dw[0][tid];
  if (tid < 80)  dw_s[48 + tid]  = P.dw[1][tid];
  if (tid < 112) dw_s[128 + tid] = P.dw[2][tid];
  if (tid < 192) {
    int br = tid >> 6, o = tid & 63;
    float sc = P.bg[br][o] * rsqrtf(P.bv[br][o] + EPSV);
    bsc[tid] = sc;
    bsh[tid] = P.bbt[br][o] - P.bm[br][o] * sc;
  }
  if (tid < 16) {
    float sc = P.ing[tid] * rsqrtf(P.invv[tid] + EPSV);
    isc[tid] = sc;
    ish[tid] = P.inb[tid] - P.inm[tid] * sc;
  }
  __syncthreads();

  const float* xb = P.x + (size_t)b * 16 * 16384;
  for (int idx = tid; idx < 16 * 260; idx += 256) {
    int c = idx / 260, i = idx - c * 260;
    int xi = 4 * t0 - 3 + i;
    float v = 0.f;
    if (xi >= 0 && xi < 16384) v = xb[c * 16384 + xi] * isc[c] + ish[c];
    xs[c * 264 + i] = v;
  }
  __syncthreads();

  const int tl = tid & 63;
  const int og = tid >> 6;
  const int o0 = og * 16;
  const size_t p = (size_t)b * 4096 + t0 + tl;
  _Float16* zout = z_h + p * 192;

  float zm[16];
  union { _Float16 h[16]; uint4 q[2]; } ob;

  // ---- branch 1: k=3, s=1, pool 4 ----
  #pragma unroll
  for (int o = 0; o < 16; ++o) zm[o] = -3.0e38f;
  for (int q = 0; q < 4; ++q) {
    float d[16];
    #pragma unroll
    for (int c = 0; c < 16; ++c) {
      const float* xr = xs + c * 264 + 4 * tl + q + 2;
      const float* w  = dw_s + c * 3;
      d[c] = w[0] * xr[0] + w[1] * xr[1] + w[2] * xr[2];
    }
    #pragma unroll
    for (int o = 0; o < 16; ++o) {
      const float* pr = pw_s + (o0 + o) * 16;
      float y = 0.f;
      #pragma unroll
      for (int c = 0; c < 16; ++c) y += pr[c] * d[c];
      y = y * bsc[o0 + o] + bsh[o0 + o];
      y = (y >= 0.f) ? y : 0.2f * y;
      zm[o] = fmaxf(zm[o], y);
    }
  }
  #pragma unroll
  for (int o = 0; o < 16; ++o) ob.h[o] = (_Float16)zm[o];
  *(uint4*)(zout + o0)     = ob.q[0];
  *(uint4*)(zout + o0 + 8) = ob.q[1];

  // ---- branch 2: k=5, s=2, pool 2 ----
  #pragma unroll
  for (int o = 0; o < 16; ++o) zm[o] = -3.0e38f;
  for (int q = 0; q < 2; ++q) {
    float d[16];
    #pragma unroll
    for (int c = 0; c < 16; ++c) {
      const float* xr = xs + c * 264 + 4 * tl + 2 * q + 1;
      const float* w  = dw_s + 48 + c * 5;
      d[c] = w[0]*xr[0] + w[1]*xr[1] + w[2]*xr[2] + w[3]*xr[3] + w[4]*xr[4];
    }
    #pragma unroll
    for (int o = 0; o < 16; ++o) {
      const float* pr = pw_s + 1024 + (o0 + o) * 16;
      float y = 0.f;
      #pragma unroll
      for (int c = 0; c < 16; ++c) y += pr[c] * d[c];
      y = y * bsc[64 + o0 + o] + bsh[64 + o0 + o];
      y = (y >= 0.f) ? y : 0.2f * y;
      zm[o] = fmaxf(zm[o], y);
    }
  }
  #pragma unroll
  for (int o = 0; o < 16; ++o) ob.h[o] = (_Float16)zm[o];
  *(uint4*)(zout + 64 + o0)     = ob.q[0];
  *(uint4*)(zout + 64 + o0 + 8) = ob.q[1];

  // ---- branch 3: k=7, s=4, pool 1 ----
  {
    float d[16];
    #pragma unroll
    for (int c = 0; c < 16; ++c) {
      const float* xr = xs + c * 264 + 4 * tl;
      const float* w  = dw_s + 128 + c * 7;
      d[c] = w[0]*xr[0] + w[1]*xr[1] + w[2]*xr[2] + w[3]*xr[3]
           + w[4]*xr[4] + w[5]*xr[5] + w[6]*xr[6];
    }
    #pragma unroll
    for (int o = 0; o < 16; ++o) {
      const float* pr = pw_s + 2048 + (o0 + o) * 16;
      float y = 0.f;
      #pragma unroll
      for (int c = 0; c < 16; ++c) y += pr[c] * d[c];
      y = y * bsc[128 + o0 + o] + bsh[128 + o0 + o];
      y = (y >= 0.f) ? y : 0.2f * y;
      ob.h[o] = (_Float16)y;
    }
  }
  *(uint4*)(zout + 128 + o0)     = ob.q[0];
  *(uint4*)(zout + 128 + o0 + 8) = ob.q[1];
}

// ---------------------------------------------------------------- kernel W --
__global__ __launch_bounds__(256) void kernW(const float* __restrict__ w,
                                             _Float16* __restrict__ w_h,
                                             const float* g, const float* be,
                                             const float* m, const float* v,
                                             float* __restrict__ oscale,
                                             float* __restrict__ oshift) {
  int tid = threadIdx.x;
  if (blockIdx.x < 192) {
    int i = blockIdx.x * 256 + tid;
    w_h[i] = (_Float16)w[i];
  } else {
    float sc = g[tid] * rsqrtf(v[tid] + EPSV);
    oscale[tid] = sc;
    oshift[tid] = be[tid] - m[tid] * sc;
  }
}

// ---------------------------------------------------------------- kernel B --
// z_f[b,o,t] = BN( sum_c out_w[o,c] * z[b,c,t] ) via v_wmma_f32_16x16x32_f16
// one wave = one 16-position tile, loops all 16 M-tiles, B kept in VGPRs.
__global__ __launch_bounds__(256) void kernB(const _Float16* __restrict__ z_h,
                                             const _Float16* __restrict__ w_h,
                                             const float* __restrict__ oscale,
                                             const float* __restrict__ oshift,
                                             float* __restrict__ zf) {
  const int lane = threadIdx.x & 31;
  const int wave = threadIdx.x >> 5;
  const int ntile = blockIdx.x * 8 + wave;
  const int p0 = ntile * 16;
  const int b  = p0 >> 12;
  const int t0 = p0 & 4095;
  const int nl = lane & 15;
  const int kh = lane >> 4;

  // B tiles: 32x16 f16, lane = column (t), kh selects K octet; K [8kh,8kh+8) and [16+8kh,24+8kh)
  H16 Bt[6];
  const _Float16* zr = z_h + (size_t)(p0 + nl) * 192 + kh * 8;
  #pragma unroll
  for (int ki = 0; ki < 6; ++ki) {
    const _Float16* ptr = zr + ki * 32;
    Bt[ki].q[0] = *(const uint4*)ptr;
    Bt[ki].q[1] = *(const uint4*)(ptr + 16);
  }

  for (int m = 0; m < 16; ++m) {
    v8f acc = {0.f, 0.f, 0.f, 0.f, 0.f, 0.f, 0.f, 0.f};
    const _Float16* wr = w_h + (size_t)(m * 16 + nl) * 192 + kh * 8;
    #pragma unroll
    for (int ki = 0; ki < 6; ++ki) {
      H16 At;                                   // 16x32 f16 A tile
      At.q[0] = *(const uint4*)(wr + ki * 32);        // K [kbase, kbase+8)
      At.q[1] = *(const uint4*)(wr + ki * 32 + 16);   // K [kbase+16, kbase+24)
      acc = __builtin_amdgcn_wmma_f32_16x16x32_f16(
          false, At.v, false, Bt[ki].v, (short)0, acc, false, false);
    }
    const int ob = m * 16 + kh * 8;             // VGPR j -> row ob+j
    float* out = zf + ((size_t)(b * 256 + ob)) * 4096 + t0 + nl;
    #pragma unroll
    for (int j = 0; j < 8; ++j)
      out[(size_t)j * 4096] = acc[j] * oscale[ob + j] + oshift[ob + j];
  }
}

// ---------------------------------------------------------------- kernel C --
__global__ __launch_bounds__(256) void kernC(const float* __restrict__ zf,
                                             float* __restrict__ s) {
  __shared__ float red[256];
  const int row = blockIdx.x, tid = threadIdx.x;
  const float* src = zf + (size_t)row * 4096;
  float sum = 0.f;
  for (int i = tid; i < 4096; i += 256) sum += src[i];
  red[tid] = sum; __syncthreads();
  for (int off = 128; off > 0; off >>= 1) {
    if (tid < off) red[tid] += red[tid + off];
    __syncthreads();
  }
  if (tid == 0) s[row] = red[0] * (1.0f / 4096.0f);
}

// ---------------------------------------------------------------- kernel D --
struct DLayer {
  const float *in_w, *in_b, *out_w, *out_b;
  const float *ff1_w, *ff1_b, *ff2_w, *ff2_b;
  const float *ln1_g, *ln1_b, *ln2_g, *ln2_b;
};
struct DParams {
  const float *W, *bvec, *alpha;
  const float *Wu, *bu, *Ws, *bs, *W1, *b1, *W2, *b2;
  DLayer L[2];
};

__device__ __forceinline__ float block_sum(float v, float* red, int tid) {
  red[tid] = v; __syncthreads();
  for (int off = 128; off > 0; off >>= 1) {
    if (tid < off) red[tid] += red[tid + off];
    __syncthreads();
  }
  float r = red[0]; __syncthreads();
  return r;
}

__global__ __launch_bounds__(256) void kernD(DParams P,
                                             const float* __restrict__ s_in,
                                             float* __restrict__ h_out) {
  __shared__ float sv[256], uv[256], hv[256], t1[2048], t2[256], red[256];
  const int b = blockIdx.x, tid = threadIdx.x;

  float s0 = s_in[b * 256 + tid];
  sv[tid] = s0; uv[tid] = s0;
  __syncthreads();

  // unfold scan: u = relu(u - a*(u@W^T + b))
  for (int k = 0; k < 10; ++k) {
    float a = P.alpha[k];
    const float* wr = P.W + (size_t)tid * 256;
    float dot = P.bvec[tid];
    for (int j = 0; j < 256; ++j) dot += uv[j] * wr[j];
    __syncthreads();
    uv[tid] = fmaxf(uv[tid] - a * dot, 0.f);
    __syncthreads();
  }

  // fusion gate
  float pdot = P.bu[tid];
  { const float* wr = P.Wu + (size_t)tid * 256;
    for (int j = 0; j < 256; ++j) pdot += uv[j] * wr[j]; }
  float qdot = P.bs[tid];
  { const float* wr = P.Ws + (size_t)tid * 256;
    for (int j = 0; j < 256; ++j) qdot += sv[j] * wr[j]; }
  t2[tid] = qdot + sv[tid];
  __syncthreads();
  if (tid < 128) {
    const float* wr = P.W1 + (size_t)tid * 256;
    float r = P.b1[tid];
    for (int j = 0; j < 256; ++j) r += t2[j] * wr[j];
    t1[tid] = fmaxf(r, 0.f);
  }
  __syncthreads();
  float a2 = P.b2[tid];
  { const float* wr = P.W2 + (size_t)tid * 128;
    for (int j = 0; j < 128; ++j) a2 += t1[j] * wr[j]; }
  float al = 1.f / (1.f + expf(-a2));
  hv[tid] = al * pdot + (1.f - al) * sv[tid];
  __syncthreads();

  // 2 "attention" layers (per-row linears + LN)
  for (int l = 0; l < 2; ++l) {
    const DLayer& L = P.L[l];
    float v = L.in_b[512 + tid];
    { const float* wr = L.in_w + (size_t)(512 + tid) * 256;
      for (int j = 0; j < 256; ++j) v += hv[j] * wr[j]; }
    t2[tid] = v; __syncthreads();
    float at = L.out_b[tid];
    { const float* wr = L.out_w + (size_t)tid * 256;
      for (int j = 0; j < 256; ++j) at += t2[j] * wr[j]; }
    float x2 = hv[tid] + at;
    float mu  = block_sum(x2, red, tid) * (1.f / 256.f);
    float var = block_sum((x2 - mu) * (x2 - mu), red, tid) * (1.f / 256.f);
    hv[tid] = (x2 - mu) * rsqrtf(var + EPSV) * L.ln1_g[tid] + L.ln1_b[tid];
    __syncthreads();

    for (int f = tid; f < 2048; f += 256) {
      const float* wr = L.ff1_w + (size_t)f * 256;
      float r = L.ff1_b[f];
      for (int j = 0; j < 256; ++j) r += hv[j] * wr[j];
      t1[f] = fmaxf(r, 0.f);
    }
    __syncthreads();
    float f2 = L.ff2_b[tid];
    { const float* wr = L.ff2_w + (size_t)tid * 2048;
      for (int j = 0; j < 2048; ++j) f2 += t1[j] * wr[j]; }
    x2 = hv[tid] + f2;
    mu  = block_sum(x2, red, tid) * (1.f / 256.f);
    var = block_sum((x2 - mu) * (x2 - mu), red, tid) * (1.f / 256.f);
    float hn = (x2 - mu) * rsqrtf(var + EPSV) * L.ln2_g[tid] + L.ln2_b[tid];
    __syncthreads();
    hv[tid] = hn;
    __syncthreads();
  }
  h_out[b * 256 + tid] = hv[tid];
}

// --------------------------------------------------------------- launcher ---
extern "C" void kernel_launch(void* const* d_in, const int* in_sizes, int n_in,
                              void* d_out, int out_size, void* d_ws, size_t ws_size,
                              hipStream_t stream) {
  (void)in_sizes; (void)n_in; (void)out_size; (void)ws_size;
  const float* F(int i);
  #define IN(i) ((const float*)d_in[(i)])

  // workspace layout
  char* ws = (char*)d_ws;
  _Float16* z_h    = (_Float16*)ws;                         // 131072*192*2 = 50331648
  _Float16* w_h    = (_Float16*)(ws + 50331648);            // 98304
  float*    oscale = (float*)(ws + 50429952);               // 1024
  float*    oshift = (float*)(ws + 50430976);               // 1024
  float*    s_ws   = (float*)(ws + 50432000);               // 32768

  float* out   = (float*)d_out;
  float* h_out = out;            // (32,256)
  float* zf    = out + 8192;     // (32,256,4096)

  // ---- kernel A params (input indices per setup_inputs dict order) ----
  AParams A;
  A.x = IN(0);
  A.ing = IN(1); A.inb = IN(2); A.inm = IN(3); A.invv = IN(4);
  // b1: 5..10, b2: 11..16, b3: 17..22
  A.dw[0] = IN(5);  A.pw[0] = IN(6);
  A.bg[0] = IN(7);  A.bbt[0] = IN(8);  A.bm[0] = IN(9);  A.bv[0] = IN(10);
  A.dw[1] = IN(11); A.pw[1] = IN(12);
  A.bg[1] = IN(13); A.bbt[1] = IN(14); A.bm[1] = IN(15); A.bv[1] = IN(16);
  A.dw[2] = IN(17); A.pw[2] = IN(18);
  A.bg[2] = IN(19); A.bbt[2] = IN(20); A.bm[2] = IN(21); A.bv[2] = IN(22);

  kernA<<<2048, 256, 0, stream>>>(A, z_h);
  kernW<<<193, 256, 0, stream>>>(IN(23), w_h, IN(24), IN(25), IN(26), IN(27),
                                 oscale, oshift);
  kernB<<<1024, 256, 0, stream>>>(z_h, w_h, oscale, oshift, zf);
  kernC<<<8192, 256, 0, stream>>>(zf, s_ws);

  DParams D;
  D.W = IN(28); D.bvec = IN(29); D.alpha = IN(30);
  D.Wu = IN(31); D.bu = IN(32); D.Ws = IN(33); D.bs = IN(34);
  D.W1 = IN(35); D.b1 = IN(36); D.W2 = IN(37); D.b2 = IN(38);
  for (int l = 0; l < 2; ++l) {
    int base = 39 + l * 12;
    D.L[l].in_w  = IN(base + 0);  D.L[l].in_b  = IN(base + 1);
    D.L[l].out_w = IN(base + 2);  D.L[l].out_b = IN(base + 3);
    D.L[l].ff1_w = IN(base + 4);  D.L[l].ff1_b = IN(base + 5);
    D.L[l].ff2_w = IN(base + 6);  D.L[l].ff2_b = IN(base + 7);
    D.L[l].ln1_g = IN(base + 8);  D.L[l].ln1_b = IN(base + 9);
    D.L[l].ln2_g = IN(base + 10); D.L[l].ln2_b = IN(base + 11);
  }
  kernD<<<32, 256, 0, stream>>>(D, s_ws, h_out);
  #undef IN
}
